// EDCFlowNet_47966194762266
// MI455X (gfx1250) — matrix operand order repaired and checked by reading, not compiled, not run on hardware
//
#include <hip/hip_runtime.h>
#include <math.h>

// ---------------------------------------------------------------------------
// EDCFlow forward on gfx1250.  All convs + correlation run on
// v_wmma_f32_16x16x32_f16.  Weights pre-packed to f16 A-fragment layout
// (M-tiles padded to a multiple of 4 so the WMMA is unconditional); im2col
// B tile double-buffered in LDS, shared by 4 waves (one barrier per k-step);
// gather is fully branchless (clamped address + multiply mask).
// ---------------------------------------------------------------------------

#define MAXK 2592   // largest Cin*KH*KW (GRU convs: 288*9); multiple of 32

typedef _Float16 half16 __attribute__((ext_vector_type(16)));
typedef _Float16 half8v __attribute__((ext_vector_type(8)));
typedef _Float16 half4v __attribute__((ext_vector_type(4)));
typedef float    float8 __attribute__((ext_vector_type(8)));

static __device__ __forceinline__ int kmapA(int lane, int h) {
  // 16-bit A-matrix 16x32 layout (ISA 7.12.2): lanes0-15 K={0..7,16..23},
  // lanes16-31 K={8..15,24..31}; 2 halves per VGPR.
  return ((lane & 16) ? 8 : 0) + ((h & 8) ? 16 : 0) + (h & 7);
}
static __device__ __forceinline__ int iclampi(int v, int lo, int hi) {
  return v < lo ? lo : (v > hi ? hi : v);
}
static __device__ __forceinline__ float tap2d(const float* img, int W, int H,
                                              int xi, int yi) {
  if (xi < 0 || xi >= W || yi < 0 || yi >= H) return 0.f;
  return img[yi * W + xi];
}
// branchless activations (v_exp_f32 path, no libm branches)
static __device__ __forceinline__ float fast_tanh(float x) {
  x = fminf(fmaxf(x, -15.f), 15.f);
  float e = __expf(2.f * x);
  return (e - 1.f) / (e + 1.f);
}
static __device__ __forceinline__ float fast_sigmoid(float x) {
  x = fminf(fmaxf(x, -30.f), 30.f);
  return 1.f / (1.f + __expf(-x));
}

// ---------------------------------------------------------------------------
// Weight pre-pack: f32 OIHW -> f16 fragments [mtile][kb][lane][16]
// grid = (KB*2, MtPadded); 256 threads; rows m>=Cout zero-filled.
// ---------------------------------------------------------------------------
__global__ void pack_wgt_kernel(const float* __restrict__ w,
                                _Float16* __restrict__ out,
                                int Cout, int K, int KB)
{
  int e  = blockIdx.x * 256 + threadIdx.x;   // 0 .. KB*512-1
  int mt = blockIdx.y;
  int h    = e & 15;
  int lane = (e >> 4) & 31;
  int kb   = e >> 9;
  int m  = mt * 16 + (lane & 15);
  int ka = kb * 32 + kmapA(lane, h);
  float v = (m < Cout && ka < K) ? w[(size_t)m * K + ka] : 0.f;
  out[((size_t)mt * KB) * 512 + e] = (_Float16)v;
}

// feature transpose+convert: f32 [C][NP] -> f16 [NP][C]; grid=(NP/256, C)
__global__ void pack_featT_kernel(const float* __restrict__ src,
                                  _Float16* __restrict__ dst, int C, int NP)
{
  int p = blockIdx.x * 256 + threadIdx.x;
  int c = blockIdx.y;
  if (p >= NP) return;
  dst[(size_t)p * C + c] = (_Float16)src[(size_t)c * NP + p];
}

// ---------------------------------------------------------------------------
// Implicit-GEMM conv.  Block = 128 threads = 4 waves.
//   blockIdx.x = n-tile (16 output positions, double-buffered B tile in LDS)
//   blockIdx.y*4 + wave = m-tile (padded; weights zero-filled beyond Cout)
// act: 0=none 1=relu 2=sigmoid 3=tanh
// ---------------------------------------------------------------------------
__global__ __launch_bounds__(128)
void wmma_conv_kernel(const float* __restrict__ in,
                      const _Float16* __restrict__ wp,
                      const float* __restrict__ bias, float* __restrict__ out,
                      int Cin, int Cout, int IH, int IW, int KH, int KW,
                      int stride, int pad, int OH, int OW,
                      int act, float postScale, int inImgStride,
                      int outImgStride)
{
  __shared__ __align__(32) _Float16 ldsB[2][16 * 32];  // [n][k], ping-pong
  __shared__ int dtab[MAXK];                           // k -> (ci,kh,kw)

  const int tid  = threadIdx.x;
  const int lane = tid & 31;
  const int wave = tid >> 5;
  const int NP = OH * OW;
  const int ntile = blockIdx.x;
  const int mtile = blockIdx.y * 4 + wave;   // always valid (padded weights)
  const int img = blockIdx.z;

  const float* inp  = in  + (size_t)img * inImgStride;
  float*       outp = out + (size_t)img * outImgStride;

  const int K   = Cin * KH * KW;
  const int KHW = KH * KW;
  const int KB  = (K + 31) >> 5;
  const int Kpad = KB * 32;                  // <= MAXK

  // decode table (ci,kh,kw); padded to Kpad so the gather needs no clamp
  for (int k = tid; k < Kpad; k += 128) {
    int kc = k < K ? k : K - 1;
    int ci = kc / KHW;
    int r  = kc - ci * KHW;
    int kh = r / KW;
    dtab[k] = (ci << 16) | (kh << 8) | (r - kh * KW);
  }

  // LDS-fill assignment: thread writes 4 consecutive k for one n
  const int fn = tid >> 3;         // 0..15
  const int fk = (tid & 7) * 4;    // 0..28
  const int fp = ntile * 16 + fn;
  const bool fvalid = fp < NP;
  const int fpp = fvalid ? fp : NP - 1;
  const int foy = fpp / OW, fox = fpp - (fpp / OW) * OW;
  const int fby = foy * stride - pad;
  const int fbx = fox * stride - pad;

  // fully branchless gather: clamped (always-valid) address, multiply mask
  auto gather = [&](int kb) -> half4v {
    half4v t;
    int kbase = kb * 32 + fk;
#pragma unroll
    for (int j = 0; j < 4; ++j) {
      int kk = kbase + j;
      int e  = dtab[kk];                     // 4 consecutive -> ds_load_b128
      int ci = e >> 16, kh = (e >> 8) & 255, kw = e & 255;
      int iy = fby + kh, ix = fbx + kw;
      int iyc = iclampi(iy, 0, IH - 1);
      int ixc = iclampi(ix, 0, IW - 1);
      float v = inp[((size_t)ci * IH + iyc) * IW + ixc];  // always in-range
      float okf = (fvalid && kk < K && iy == iyc && ix == ixc) ? 1.f : 0.f;
      t[j] = (_Float16)(v * okf);            // v consumed unconditionally
    }
    return t;
  };

  // wmma lane state
  const int n = lane & 15;
  const int p = ntile * 16 + n;
  const bool pvalid = p < NP;
  const _Float16* wtile = wp + (size_t)mtile * KB * 512 + (size_t)lane * 16;

  float8 acc = {0.f, 0.f, 0.f, 0.f, 0.f, 0.f, 0.f, 0.f};

  __syncthreads();                 // dtab ready
  *(half4v*)&ldsB[0][fn * 32 + fk] = gather(0);
  __syncthreads();                 // tile 0 ready

  for (int kb = 0; kb < KB; ++kb) {
    if (kb + 1 < KB)               // prefetch next tile into other buffer
      *(half4v*)&ldsB[(kb + 1) & 1][fn * 32 + fk] = gather(kb + 1);
    half16 a = *(const half16*)(wtile + (size_t)kb * 512);
    const _Float16* brow = &ldsB[kb & 1][(lane & 15) * 32 + ((lane & 16) >> 1)];
    half16 b = __builtin_shufflevector(*(const half8v*)brow, *(const half8v*)(brow + 16),
                                       0, 1, 2, 3, 4, 5, 6, 7, 8, 9, 10, 11, 12, 13, 14, 15);
    acc = __builtin_amdgcn_wmma_f32_16x16x32_f16(false, a, false, b,
                                                 (short)0, acc, false, false);
    __syncthreads();               // next tile ready / current reads done
  }

  // D layout: vgpr i -> row mtile*16 + i + 8*(lane>=16), col = lane&15
  const int mbase = mtile * 16 + ((lane & 16) ? 8 : 0);
#pragma unroll
  for (int i = 0; i < 8; ++i) {
    int mo = mbase + i;
    if (mo < Cout && pvalid) {
      float v = (acc[i] + bias[mo]) * postScale;
      if (act == 1)      v = fmaxf(v, 0.f);
      else if (act == 2) v = fast_sigmoid(v);
      else if (act == 3) v = fast_tanh(v);
      outp[(size_t)mo * NP + p] = v;
    }
  }
}

// ---------------------------------------------------------------------------
// corr[m][n] = scale * sum_c f0[m][c] * fT[n][c]   (features pre-transposed
// to f16 [pos][C]); M=N=2240, C=128.
// ---------------------------------------------------------------------------
__global__ __launch_bounds__(128)
void wmma_corr_kernel(const _Float16* __restrict__ f0p,
                      const _Float16* __restrict__ fTp,
                      float* __restrict__ corr, int NP, int C, float scale)
{
  const int lane = threadIdx.x & 31;
  const int wave = threadIdx.x >> 5;
  const int ntile = blockIdx.x * 4 + wave;
  if (ntile >= (NP >> 4)) return;
  const int mtile = blockIdx.y;
  const int m = mtile * 16 + (lane & 15);
  const int n = ntile * 16 + (lane & 15);
  const int a0 = (lane & 16) ? 8 : 0;

  float8 acc = {0.f, 0.f, 0.f, 0.f, 0.f, 0.f, 0.f, 0.f};
  for (int kb = 0; kb < (C >> 5); ++kb) {
    const _Float16* arow = f0p + (size_t)m * C + kb * 32;
    half8v alo = *(const half8v*)(arow + a0);        // K = base+0..7
    half8v ahi = *(const half8v*)(arow + a0 + 16);   // K = base+16..23
    half16 a = __builtin_shufflevector(alo, ahi, 0, 1, 2, 3, 4, 5, 6, 7,
                                       8, 9, 10, 11, 12, 13, 14, 15);
    const _Float16* brow = fTp + (size_t)n * C + kb * 32 + a0;
    half16 b = __builtin_shufflevector(*(const half8v*)brow, *(const half8v*)(brow + 16),
                                       0, 1, 2, 3, 4, 5, 6, 7, 8, 9, 10, 11, 12, 13, 14, 15);
    acc = __builtin_amdgcn_wmma_f32_16x16x32_f16(false, a, false, b,
                                                 (short)0, acc, false, false);
  }
  const int mbase = mtile * 16 + ((lane & 16) ? 8 : 0);
#pragma unroll
  for (int i = 0; i < 8; ++i)
    corr[(size_t)(mbase + i) * NP + n] = acc[i] * scale;
}

// ---------------------------------------------------------------------------
// Scalar glue kernels (2D grids: blockIdx.y = channel/tap, no int division)
// ---------------------------------------------------------------------------
__global__ void pack_frames_kernel(const float* __restrict__ x1,
                                   const float* __restrict__ x2,
                                   float* __restrict__ frames, int HW)
{
  int p = blockIdx.x * 256 + threadIdx.x;
  int c = blockIdx.y;                       // 0..17
  if (p >= HW) return;
  frames[(size_t)c * HW + p] =
      (c < 3) ? x1[(size_t)(12 + c) * HW + p] : x2[(size_t)(c - 3) * HW + p];
}

__global__ void cnet_split_kernel(const float* __restrict__ cmap,
                                  float* __restrict__ netb,
                                  float* __restrict__ hx, int NP)
{
  int p = blockIdx.x * 256 + threadIdx.x;
  int c = blockIdx.y;                       // 0..159
  if (p >= NP) return;
  float v = cmap[(size_t)c * NP + p];
  if (c < 96) netb[(size_t)c * NP + p] = fast_tanh(v);
  else        hx[(size_t)(224 + (c - 96)) * NP + p] = fmaxf(v, 0.f);  // inp4
}

__global__ void init_coords_kernel(float* __restrict__ crd, int H, int W)
{
  int NP = H * W;
  int p = blockIdx.x * 256 + threadIdx.x;
  int c = blockIdx.y;                       // 0..1
  if (p >= NP) return;
  crd[(size_t)c * NP + p] = (c == 0) ? (float)(p % W) : (float)(p / W);
}

__global__ void flow_kernel(const float* __restrict__ crd,
                            float* __restrict__ flb, float* __restrict__ hx,
                            int H, int W)
{
  int NP = H * W;
  int p = blockIdx.x * 256 + threadIdx.x;
  int c = blockIdx.y;                       // 0..1
  if (p >= NP) return;
  float g = (c == 0) ? (float)(p % W) : (float)(p / W);
  float v = crd[(size_t)c * NP + p] - g;
  flb[(size_t)c * NP + p] = v;
  hx[(size_t)(222 + c) * NP + p] = v;       // flow channels of x
}

__global__ void corr_lookup_kernel(const float* __restrict__ corr,
                                   const float* __restrict__ crd,
                                   float* __restrict__ cms,
                                   int H8, int W8, int H4, int W4)
{
  int NP8 = H8 * W8, NP4 = H4 * W4;
  int p = blockIdx.x * 256 + threadIdx.x;
  int j = blockIdx.y;                       // 0..48 tap index
  if (p >= NP8) return;
  int h8 = p / W8, w8 = p - h8 * W8;
  float cx = 0.f, cy = 0.f;
  for (int dy = 0; dy < 2; ++dy)
    for (int dx = 0; dx < 2; ++dx) {
      int q = (h8 * 2 + dy) * W4 + (w8 * 2 + dx);
      cx += crd[q];
      cy += crd[NP4 + q];
    }
  cx *= 0.125f; cy *= 0.125f;               // 2x2 mean then /2
  float xs = cx + (float)(j % 7 - 3);
  float ys = cy + (float)(j / 7 - 3);
  float x0 = floorf(xs), y0 = floorf(ys);
  float wx = xs - x0, wy = ys - y0;
  int x0i = (int)x0, y0i = (int)y0;
  const float* row = corr + (size_t)p * NP8;
  float v = tap2d(row, W8, H8, x0i,     y0i    ) * (1.f - wx) * (1.f - wy)
          + tap2d(row, W8, H8, x0i + 1, y0i    ) * wx         * (1.f - wy)
          + tap2d(row, W8, H8, x0i,     y0i + 1) * (1.f - wx) * wy
          + tap2d(row, W8, H8, x0i + 1, y0i + 1) * wx         * wy;
  cms[(size_t)j * NP8 + p] = v;
}

__global__ void resize_lin_kernel(const float* __restrict__ src,
                                  float* __restrict__ dst,
                                  int IH, int IW, int OH, int OW)
{
  int NPo = OH * OW;
  int p = blockIdx.x * 256 + threadIdx.x;
  int c = blockIdx.y;
  if (p >= NPo) return;
  int oy = p / OW, ox = p - (p / OW) * OW;
  float sy = (oy + 0.5f) * ((float)IH / OH) - 0.5f;
  float sx = (ox + 0.5f) * ((float)IW / OW) - 0.5f;
  int y0 = (int)floorf(sy); float fy = sy - (float)y0;
  int x0 = (int)floorf(sx); float fx = sx - (float)x0;
  int y0c = iclampi(y0, 0, IH - 1), y1c = iclampi(y0 + 1, 0, IH - 1);
  int x0c = iclampi(x0, 0, IW - 1), x1c = iclampi(x0 + 1, 0, IW - 1);
  const float* s = src + (size_t)c * IH * IW;
  dst[(size_t)c * NPo + p] =
      s[y0c * IW + x0c] * (1.f - fy) * (1.f - fx)
    + s[y0c * IW + x1c] * (1.f - fy) * fx
    + s[y1c * IW + x0c] * fy * (1.f - fx)
    + s[y1c * IW + x1c] * fy * fx;
}

__global__ void warp_mean_kernel(const float* __restrict__ f4,
                                 const float* __restrict__ crd,
                                 float* __restrict__ wmean,
                                 int C, int H, int W, int T)
{
  int NP = H * W;
  int p = blockIdx.x * 256 + threadIdx.x;
  int c = blockIdx.y;                       // 0..C-1
  if (p >= NP) return;
  int h = p / W, w = p - h * W;
  float fx = crd[p] - (float)w;
  float fy = crd[NP + p] - (float)h;
  float acc = f4[(size_t)c * NP + p];       // frame 0 unwarped
  float sxw = (float)W / (float)(W - 1);
  float syw = (float)H / (float)(H - 1);
  for (int t = 1; t < T; ++t) {
    float s = (float)t / (float)(T - 1);
    float px = ((float)w + fx * s) * sxw - 0.5f;
    float py = ((float)h + fy * s) * syw - 0.5f;
    float x0 = floorf(px), y0 = floorf(py);
    float wx = px - x0, wy = py - y0;
    int x0i = (int)x0, y0i = (int)y0;
    const float* img = f4 + (size_t)t * C * NP + (size_t)c * NP;
    acc += tap2d(img, W, H, x0i,     y0i    ) * (1.f - wx) * (1.f - wy)
         + tap2d(img, W, H, x0i + 1, y0i    ) * wx         * (1.f - wy)
         + tap2d(img, W, H, x0i,     y0i + 1) * (1.f - wx) * wy
         + tap2d(img, W, H, x0i + 1, y0i + 1) * wx         * wy;
  }
  wmean[(size_t)c * NP + p] = acc / (float)T;
}

__global__ void copy_kernel(const float* __restrict__ src,
                            float* __restrict__ dst, int count)
{
  int i = blockIdx.x * 256 + threadIdx.x;
  if (i < count) dst[i] = src[i];
}

__global__ void mul_rn_kernel(const float* __restrict__ r,
                              const float* __restrict__ net,
                              float* __restrict__ dst, int count)
{
  int i = blockIdx.x * 256 + threadIdx.x;
  if (i < count) dst[i] = r[i] * net[i];
}

__global__ void gru_update_kernel(const float* __restrict__ z,
                                  const float* __restrict__ q,
                                  float* __restrict__ net, int count)
{
  int i = blockIdx.x * 256 + threadIdx.x;
  if (i < count) net[i] = (1.f - z[i]) * net[i] + z[i] * q[i];
}

__global__ void add_kernel(float* __restrict__ dst,
                           const float* __restrict__ src, int count)
{
  int i = blockIdx.x * 256 + threadIdx.x;
  if (i < count) dst[i] += src[i];
}

// convex upsample x4; grid=(NP/256, 16), blockIdx.y = di*4+dj
__global__ void upsample_kernel(const float* __restrict__ crd,
                                const float* __restrict__ mask,
                                float* __restrict__ out, int H, int W)
{
  int NP = H * W;
  int p = blockIdx.x * 256 + threadIdx.x;
  int sub = blockIdx.y;
  if (p >= NP) return;
  int h = p / W, w = p - h * W;
  int di = sub >> 2, dj = sub & 3;
  float e[9];
  float mx = -1e30f;
#pragma unroll
  for (int k = 0; k < 9; ++k) {
    float v = mask[(size_t)(k * 16 + sub) * NP + p];
    e[k] = v;
    mx = fmaxf(mx, v);
  }
  float s = 0.f;
#pragma unroll
  for (int k = 0; k < 9; ++k) { e[k] = __expf(e[k] - mx); s += e[k]; }
  float inv = 1.f / s;
  float o0 = 0.f, o1 = 0.f;
#pragma unroll
  for (int k = 0; k < 9; ++k) {
    int hh = h + k / 3 - 1;
    int ww = w + (k % 3) - 1;
    float fx = 0.f, fy = 0.f;
    if (hh >= 0 && hh < H && ww >= 0 && ww < W) {
      int q = hh * W + ww;
      fx = 4.f * (crd[q] - (float)ww);
      fy = 4.f * (crd[NP + q] - (float)hh);
    }
    float m = e[k] * inv;
    o0 += m * fx;
    o1 += m * fy;
  }
  int OW4 = W * 4;
  size_t opix = (size_t)(h * 4 + di) * OW4 + (w * 4 + dj);
  size_t NPo = (size_t)H * 4 * OW4;
  out[opix] = o0;
  out[NPo + opix] = o1;
}

// ---------------------------------------------------------------------------
// Host launcher
// ---------------------------------------------------------------------------
static inline int ceil_div(int a, int b) { return (a + b - 1) / b; }

extern "C" void kernel_launch(void* const* d_in, const int* in_sizes, int n_in,
                              void* d_out, int out_size, void* d_ws, size_t ws_size,
                              hipStream_t stream)
{
  (void)in_sizes; (void)n_in; (void)out_size;
  const float* x1 = (const float*)d_in[0];
  const float* x2 = (const float*)d_in[1];
  // params flattened in jax pytree (sorted-key) order
  enum {
    CNET_B = 0, CNET_W, ENC_CORR_B, ENC_CORR_W, ENC_FLOW_B, ENC_FLOW_W,
    ENC_MOT_B, ENC_MOT_W, ENC_WARP_B, ENC_WARP_W,
    FH_B1, FH_B2, FH_W1, FH_W2, FNET_B1, FNET_B2, FNET_W1, FNET_W2,
    GRU_BQ, GRU_BR, GRU_BZ, GRU_WQ, GRU_WR, GRU_WZ,
    MH_B1, MH_B2, MH_W1, MH_W2, NPARAM
  };
  const float* prm[NPARAM];
  for (int i = 0; i < NPARAM; ++i) prm[i] = (const float*)d_in[2 + i];
  const int iters = 4;  // device scalar d_in[30] unreadable under graph capture

  const int H = 320, W = 448, HW = H * W;
  const int H4 = 80, W4 = 112, NP4 = H4 * W4;     // 8960
  const int H8 = 40, W8 = 56, NP8 = H8 * W8;      // 2240
  const int T = 6;

  // workspace layout (64-byte aligned allocations)
  float* ws = (float*)d_ws;
  size_t off = 0;
  auto take = [&](size_t n) {
    n = (n + 15) & ~15ull;
    float* q = ws + off; off += n; return q;
  };
  auto takeH = [&](size_t nh) {
    size_t nf = ((nh + 1) / 2 + 15) & ~15ull;
    _Float16* q = (_Float16*)(ws + off); off += nf; return q;
  };
  float* frames = take(18ull * HW);           // (6,3,320,448) == (18,320,448)
  float* f4     = take(6ull * 64 * NP4);
  float* f8b    = take(6ull * 128 * NP8);
  float* cmap   = take(160ull * NP4);
  float* netb   = take(96ull * NP4);
  float* corr5  = take((size_t)NP8 * NP8);    // 20 MB
  float* cms    = take(49ull * NP8);
  float* cm     = take(49ull * NP4);
  float* wmean  = take(64ull * NP4);
  float* catCFW = take(192ull * NP4);         // [cfeat|ffeat|wfeat]
  float* hx     = take(288ull * NP4);         // [net|mot|flow|inp4]
  float* rx     = take(288ull * NP4);         // [r*net | x]
  float* zb     = take(96ull * NP4);
  float* rb     = take(96ull * NP4);
  float* qb     = take(96ull * NP4);
  float* t128   = take(128ull * NP4);
  float* dfl    = take(2ull * NP4);
  float* maskb  = take(144ull * NP4);
  float* crd    = take(2ull * NP4);
  float* flb    = take(2ull * NP4);
  _Float16* f0p = takeH((size_t)NP8 * 128);   // f16 [pos][C] frame 0
  _Float16* fTp = takeH((size_t)NP8 * 128);   // f16 [pos][C] frame 5

  // pre-pack all conv weights to f16 A-fragment layout, M-tiles padded to 4
  auto packW = [&](const float* w, int Cout, int K) -> const _Float16* {
    int MtP = ceil_div(ceil_div(Cout, 16), 4) * 4;
    int KB  = ceil_div(K, 32);
    _Float16* dst = takeH((size_t)MtP * KB * 512);
    pack_wgt_kernel<<<dim3(KB * 2, MtP), dim3(256), 0, stream>>>(w, dst, Cout, K, KB);
    return dst;
  };
  const _Float16* W_fnet1   = packW(prm[FNET_W1], 64, 48);
  const _Float16* W_fnet2   = packW(prm[FNET_W2], 128, 256);
  const _Float16* W_cnet    = packW(prm[CNET_W], 160, 288);
  const _Float16* W_enccorr = packW(prm[ENC_CORR_W], 96, 49);
  const _Float16* W_encflow = packW(prm[ENC_FLOW_W], 64, 18);
  const _Float16* W_encwarp = packW(prm[ENC_WARP_W], 32, 576);
  const _Float16* W_encmot  = packW(prm[ENC_MOT_W], 126, 1728);
  const _Float16* W_gruz    = packW(prm[GRU_WZ], 96, 2592);
  const _Float16* W_grur    = packW(prm[GRU_WR], 96, 2592);
  const _Float16* W_gruq    = packW(prm[GRU_WQ], 96, 2592);
  const _Float16* W_fh1     = packW(prm[FH_W1], 128, 864);
  const _Float16* W_fh2     = packW(prm[FH_W2], 2, 1152);
  const _Float16* W_mh1     = packW(prm[MH_W1], 128, 864);
  const _Float16* W_mh2     = packW(prm[MH_W2], 144, 128);
  if (ws_size < off * sizeof(float)) return;  // workspace too small

  auto conv = [&](const float* in, const _Float16* wp, const float* b, float* out,
                  int Cin, int Cout, int IH, int IW, int KH, int KW,
                  int s, int pad, int OH, int OW, int act, float ps,
                  int imgs, int inStride, int outStride) {
    int NPo = OH * OW;
    int MtP = ceil_div(ceil_div(Cout, 16), 4) * 4;
    dim3 g(ceil_div(NPo, 16), MtP / 4, imgs);
    wmma_conv_kernel<<<g, dim3(128), 0, stream>>>(
        in, wp, b, out, Cin, Cout, IH, IW, KH, KW, s, pad, OH, OW, act, ps,
        inStride, outStride);
  };
#define L2D(kern, count, chans, ...) \
  kern<<<dim3(ceil_div((int)(count), 256), (chans)), dim3(256), 0, stream>>>(__VA_ARGS__)
#define L1D(kern, count, ...) \
  kern<<<dim3(ceil_div((int)(count), 256)), dim3(256), 0, stream>>>(__VA_ARGS__)

  // ---- frontend -----------------------------------------------------------
  L2D(pack_frames_kernel, HW, 18, x1, x2, frames, HW);
  conv(frames, W_fnet1, prm[FNET_B1], f4, 3, 64, H, W, 4, 4, 4, 0,
       H4, W4, 1, 1.f, T, 3 * HW, 64 * NP4);
  conv(f4, W_fnet2, prm[FNET_B2], f8b, 64, 128, H4, W4, 2, 2, 2, 0,
       H8, W8, 1, 1.f, T, 64 * NP4, 128 * NP8);
  conv(frames, W_cnet, prm[CNET_B], cmap, 18, 160, H, W, 4, 4, 4, 0,
       H4, W4, 0, 1.f, 1, 0, 0);
  L2D(cnet_split_kernel, NP4, 160, cmap, netb, hx, NP4);
  // correlation volume
  L2D(pack_featT_kernel, NP8, 128, f8b, f0p, 128, NP8);
  L2D(pack_featT_kernel, NP8, 128, f8b + 5ull * 128 * NP8, fTp, 128, NP8);
  {
    dim3 g(ceil_div(NP8 / 16, 4), NP8 / 16, 1);
    wmma_corr_kernel<<<g, dim3(128), 0, stream>>>(
        f0p, fTp, corr5, NP8, 128, 0.08838834764831845f);
  }
  L2D(init_coords_kernel, NP4, 2, crd, H4, W4);

  // ---- refinement iterations ---------------------------------------------
  for (int it = 0; it < iters; ++it) {
    L1D(copy_kernel, 96 * NP4, netb, hx, 96 * NP4);                 // hx[0:96]=net
    L2D(flow_kernel, NP4, 2, crd, flb, hx, H4, W4);                 // flow + hx[222:224]
    L2D(corr_lookup_kernel, NP8, 49, corr5, crd, cms, H8, W8, H4, W4);
    L2D(resize_lin_kernel, NP4, 49, cms, cm, H8, W8, H4, W4);
    L2D(warp_mean_kernel, NP4, 64, f4, crd, wmean, 64, H4, W4, T);
    // motion encoder (writes pre-concatenated catCFW then mot into hx)
    conv(cm, W_enccorr, prm[ENC_CORR_B], catCFW,
         49, 96, H4, W4, 1, 1, 1, 0, H4, W4, 1, 1.f, 1, 0, 0);
    conv(flb, W_encflow, prm[ENC_FLOW_B], catCFW + 96ull * NP4,
         2, 64, H4, W4, 3, 3, 1, 1, H4, W4, 1, 1.f, 1, 0, 0);
    conv(wmean, W_encwarp, prm[ENC_WARP_B], catCFW + 160ull * NP4,
         64, 32, H4, W4, 3, 3, 1, 1, H4, W4, 1, 1.f, 1, 0, 0);
    conv(catCFW, W_encmot, prm[ENC_MOT_B], hx + 96ull * NP4,
         192, 126, H4, W4, 3, 3, 1, 1, H4, W4, 1, 1.f, 1, 0, 0);
    // GRU gates
    conv(hx, W_gruz, prm[GRU_BZ], zb, 288, 96, H4, W4, 3, 3, 1, 1,
         H4, W4, 2, 1.f, 1, 0, 0);
    conv(hx, W_grur, prm[GRU_BR], rb, 288, 96, H4, W4, 3, 3, 1, 1,
         H4, W4, 2, 1.f, 1, 0, 0);
    L1D(mul_rn_kernel, 96 * NP4, rb, netb, rx, 96 * NP4);           // rx[0:96]=r*net
    L1D(copy_kernel, 192 * NP4, hx + 96ull * NP4, rx + 96ull * NP4, 192 * NP4);
    conv(rx, W_gruq, prm[GRU_BQ], qb, 288, 96, H4, W4, 3, 3, 1, 1,
         H4, W4, 3, 1.f, 1, 0, 0);                                  // tanh
    L1D(gru_update_kernel, 96 * NP4, zb, qb, netb, 96 * NP4);
    // flow head
    conv(netb, W_fh1, prm[FH_B1], t128, 96, 128, H4, W4, 3, 3, 1, 1,
         H4, W4, 1, 1.f, 1, 0, 0);
    conv(t128, W_fh2, prm[FH_B2], dfl, 128, 2, H4, W4, 3, 3, 1, 1,
         H4, W4, 0, 1.f, 1, 0, 0);
    // mask head (0.25 * conv)
    conv(netb, W_mh1, prm[MH_B1], t128, 96, 128, H4, W4, 3, 3, 1, 1,
         H4, W4, 1, 1.f, 1, 0, 0);
    conv(t128, W_mh2, prm[MH_B2], maskb, 128, 144, H4, W4, 1, 1, 1, 0,
         H4, W4, 0, 0.25f, 1, 0, 0);
    L1D(add_kernel, 2 * NP4, crd, dfl, 2 * NP4);                    // coords1 += dflow
    L2D(upsample_kernel, NP4, 16, crd, maskb,
        (float*)d_out + (size_t)it * 2 * HW, H4, W4);
  }
#undef L1D
#undef L2D
}
